// HypergraphBackbone_84567906058913
// MI455X (gfx1250) — compile-verified, matrix operand order checked by reading, not run on hardware
//
#include <hip/hip_runtime.h>
#include <hip/hip_bf16.h>

// ---------------------------------------------------------------------------
// CascadedHyperGAT backbone for gfx1250 (MI455X).
//  - Dense projections (h@Wq, h@Wv) AND the type-gate MLP ([h,sig]@w1) run on
//    v_wmma_f32_16x16x32_f16 (16 WMMA issues total per tile pipeline).
//  - Scatter phases use L2-resident float atomics (working set << 192MB L2).
//  - h lives inside d_out column slices (output = concat([x, h1, h2])).
// ---------------------------------------------------------------------------

typedef __attribute__((ext_vector_type(16))) _Float16 v16h;
typedef __attribute__((ext_vector_type(8)))  float    v8f;

#define HEADS 4
#define HD    32
#define HID   128     // HEADS*HD
#define INDIM 128
#define QD    64
#define KMLP  192     // INDIM + QD
#define NT    4
#define ET    3
#define NLAYERS 2

// ---------------------------------------------------------------- utilities
__global__ void hg_zero_kernel(float* __restrict__ p, long n) {
    long t = blockIdx.x * (long)blockDim.x + threadIdx.x;
    long stride = (long)gridDim.x * blockDim.x;
    for (; t < n; t += stride) p[t] = 0.0f;
}

__global__ void hg_copy_x_kernel(const float* __restrict__ x,
                                 float* __restrict__ out, int ostride, int N) {
    long t = blockIdx.x * (long)blockDim.x + threadIdx.x;
    if (t >= (long)N * HID) return;
    long n = t >> 7;          // /128
    int  c = (int)(t & 127);
    out[n * (long)ostride + c] = x[t];
}

// ------------------------------------------------------- Q/V GEMM via WMMA
// Q = h @ Wq (N x 128 by 128 x 128, head-blocked cols), same for V.
// One wave -> one 16x16 output tile; grid.y*4+wave selects col tile 0..15
// (0..7 => Q, 8..15 => V). K=128 -> 4 chained v_wmma_f32_16x16x32_f16.
__global__ void __launch_bounds__(128)
hg_qv_wmma_kernel(const float* __restrict__ h, int hstride,
                  const float* __restrict__ wq, const float* __restrict__ wv,
                  float* __restrict__ Q, float* __restrict__ V, int N)
{
    const int lane = threadIdx.x & 31;
    const int wave = threadIdx.x >> 5;
    const int ct   = blockIdx.y * 4 + wave;          // 0..15
    const float* w   = (ct < 8) ? wq : wv;
    float*       dst = (ct < 8) ? Q  : V;
    const int n0 = (ct & 7) * 16;                    // col tile base in [0,128)
    const int m0 = blockIdx.x * 16;

    const int nloc = lane & 15;
    const int kb   = (lane >> 4) * 8;                // K sub-block per half-wave

    int row = m0 + nloc; if (row >= N) row = N - 1;  // clamp reads, EXEC stays full
    const int col  = n0 + nloc;
    const int head = col >> 5;
    const int wc   = col & 31;

    const float* hr = h + (size_t)row * hstride;
    const float* wp = w + (size_t)head * (INDIM * HD) + wc;
    __builtin_prefetch(hr, 0, 0);                    // global_prefetch_b8

    v8f acc = {};
    #pragma unroll
    for (int kc = 0; kc < 4; ++kc) {
        const int k0 = kc * 32 + kb;
        v16h a, b;
        #pragma unroll
        for (int j = 0; j < 8; ++j) {
            // A (16x32 f16): lane&15 = M row; halves j->K=k0+j, j+8->K=k0+16+j
            a[j]     = (_Float16)hr[k0 + j];
            a[j + 8] = (_Float16)hr[k0 + 16 + j];
            // B (32x16 f16): lane&15 = N col; same K striping (B = A-layout of B^T)
            b[j]     = (_Float16)wp[(size_t)(k0 + j) * HD];
            b[j + 8] = (_Float16)wp[(size_t)(k0 + 16 + j) * HD];
        }
        acc = __builtin_amdgcn_wmma_f32_16x16x32_f16(
            false, a, false, b, (short)0, acc, false, false);
    }

    // D (16x16 f32): lane<16 -> M=r, N=lane ; lane>=16 -> M=8+r, N=lane-16
    const int mhi = (lane >> 4) * 8;
    #pragma unroll
    for (int r = 0; r < 8; ++r) {
        const int m = m0 + mhi + r;
        if (m < N) dst[(size_t)m * HID + col] = acc[r];
    }
}

// --------------------------- type-gate MLP on WMMA: alpha_type (N, HEADS)
// z = tanh([h, tq[node_type]] @ w1 + b1)   (K=192, 32 cols per head)
// alpha = sigmoid(z @ w2 + b2)
// Block = 128 threads = 4 waves = 4 heads; wave owns one 16-node tile.
// 6 K-chunks x 2 col tiles = 12 chained WMMAs; the z@w2 dot reduces the
// D-fragments with a 16-lane shfl_xor tree (cols live across lanes).
__global__ void __launch_bounds__(128)
hg_mlp_alpha_wmma_kernel(const float* __restrict__ h, int hstride,
                         const int* __restrict__ node_types,
                         const float* __restrict__ tq,   // (HEADS, NT, QD)
                         const float* __restrict__ w1,   // (HEADS, 192, 32)
                         const float* __restrict__ b1,   // (HEADS, 32)
                         const float* __restrict__ w2,   // (HEADS, 32)
                         const float* __restrict__ b2,   // (HEADS,)
                         float* __restrict__ alpha, int N)
{
    const int lane = threadIdx.x & 31;
    const int head = threadIdx.x >> 5;               // wave == head
    const int m0   = blockIdx.x * 16;
    const int nloc = lane & 15;
    const int kb   = (lane >> 4) * 8;

    int row = m0 + nloc; if (row >= N) row = N - 1;  // clamp reads, EXEC stays full
    const float* hr = h + (size_t)row * hstride;
    const float* sg = tq + ((size_t)head * NT + node_types[row]) * QD;
    const float* wp = w1 + (size_t)head * (KMLP * 32);

    v8f acc0 = {}, acc1 = {};                        // col tiles [0,16) and [16,32)
    #pragma unroll
    for (int kc = 0; kc < 6; ++kc) {
        const int k0 = kc * 32 + kb;
        // chunks 0..3 read h (K<128), chunks 4..5 read the type-query row
        const float* asrc = (kc < 4) ? (hr + k0) : (sg + (k0 - INDIM));
        v16h a, bA, bB;
        #pragma unroll
        for (int j = 0; j < 8; ++j) {
            a[j]      = (_Float16)asrc[j];
            a[j + 8]  = (_Float16)asrc[16 + j];
            bA[j]     = (_Float16)wp[(size_t)(k0 + j) * 32 + nloc];
            bA[j + 8] = (_Float16)wp[(size_t)(k0 + 16 + j) * 32 + nloc];
            bB[j]     = (_Float16)wp[(size_t)(k0 + j) * 32 + nloc + 16];
            bB[j + 8] = (_Float16)wp[(size_t)(k0 + 16 + j) * 32 + nloc + 16];
        }
        acc0 = __builtin_amdgcn_wmma_f32_16x16x32_f16(
            false, a, false, bA, (short)0, acc0, false, false);
        acc1 = __builtin_amdgcn_wmma_f32_16x16x32_f16(
            false, a, false, bB, (short)0, acc1, false, false);
    }

    // per-lane partial of z@w2 for this lane's two columns
    const float b1a = b1[head * 32 + nloc];
    const float b1b = b1[head * 32 + nloc + 16];
    const float w2a = w2[head * 32 + nloc];
    const float w2b = w2[head * 32 + nloc + 16];
    float p[8];
    #pragma unroll
    for (int r = 0; r < 8; ++r)
        p[r] = tanhf(acc0[r] + b1a) * w2a + tanhf(acc1[r] + b1b) * w2b;

    // sum over 16 columns: lanes 0..15 hold rows 0..7, lanes 16..31 rows 8..15
    #pragma unroll
    for (int o = 8; o; o >>= 1) {
        #pragma unroll
        for (int r = 0; r < 8; ++r) p[r] += __shfl_xor(p[r], o, 16);
    }

    if (nloc == 0) {
        const float bb2 = b2[head];
        const int mhi = (lane >> 4) * 8;
        #pragma unroll
        for (int r = 0; r < 8; ++r) {
            const int m = m0 + mhi + r;
            if (m < N)
                alpha[(size_t)m * HEADS + head] = 1.0f / (1.0f + expf(-(p[r] + bb2)));
        }
    }
}

// ------------------- attention logits -> exp, per-edge softmax denominators
__global__ void hg_attn_ex_kernel(const float* __restrict__ Q,
                                  const float* __restrict__ alpha,
                                  const int* __restrict__ node_idx,
                                  const int* __restrict__ edge_idx,
                                  const int* __restrict__ edge_type,
                                  const float* __restrict__ ec,   // (HEADS, ET, 32)
                                  float* __restrict__ ex,
                                  float* __restrict__ s, long NNZ)
{
    long i = blockIdx.x * (long)blockDim.x + threadIdx.x;
    if (i >= NNZ) return;
    const int n  = node_idx[i];
    const int e  = edge_idx[i];
    const int et = edge_type[e];
    #pragma unroll
    for (int hd = 0; hd < HEADS; ++hd) {
        const float* q = Q + (size_t)n * HID + hd * HD;
        const float* c = ec + ((size_t)hd * ET + et) * HD;
        float a = 0.0f;
        #pragma unroll
        for (int j = 0; j < HD; ++j) a = fmaf(q[j], c[j], a);
        a = (a >= 0.0f) ? a : 0.2f * a;                 // leaky_relu(0.2)
        a *= alpha[(size_t)n * HEADS + hd];
        const float ev = expf(a);                       // logits O(0.1): max-shift unneeded
        ex[i * HEADS + hd] = ev;
        atomicAdd(&s[(size_t)e * HEADS + hd], ev);
    }
}

// -------------------------------- edge_feat accumulation: sum ex * V[node]
__global__ void hg_edge_accum_kernel(const float* __restrict__ ex,
                                     const float* __restrict__ V,
                                     const int* __restrict__ node_idx,
                                     const int* __restrict__ edge_idx,
                                     float* __restrict__ ef, long NNZ)
{
    long t = blockIdx.x * (long)blockDim.x + threadIdx.x;
    long i = t >> 2;
    int  hd = (int)(t & 3);
    if (i >= NNZ) return;
    const int n = node_idx[i];
    const int e = edge_idx[i];
    const float w = ex[i * HEADS + hd];
    const float* v = V + (size_t)n * HID + hd * HD;
    float* dst = ef + (size_t)e * HID + hd * HD;
    #pragma unroll
    for (int j = 0; j < HD; ++j) atomicAdd(&dst[j], w * v[j]);
}

__global__ void hg_edge_scale_kernel(float* __restrict__ ef,
                                     const float* __restrict__ s, int E)
{
    int t = blockIdx.x * blockDim.x + threadIdx.x;
    if (t >= E * HID) return;
    const int e  = t >> 7;
    const int hd = (t >> 5) & 3;
    const float sv = s[(size_t)e * HEADS + hd];
    if (sv > 0.0f) ef[t] /= sv;
}

// ------------------------- node_feat: sum attn * edge_feat[edge] into nodes
__global__ void hg_node_accum_kernel(const float* __restrict__ ex,
                                     const float* __restrict__ s,
                                     const float* __restrict__ ef,
                                     const int* __restrict__ node_idx,
                                     const int* __restrict__ edge_idx,
                                     float* __restrict__ nf, long NNZ)
{
    long t = blockIdx.x * (long)blockDim.x + threadIdx.x;
    long i = t >> 2;
    int  hd = (int)(t & 3);
    if (i >= NNZ) return;
    const int n = node_idx[i];
    const int e = edge_idx[i];
    const float sv = s[(size_t)e * HEADS + hd];
    const float attn = (sv > 0.0f) ? ex[i * HEADS + hd] / sv : 0.0f;
    const float* src = ef + (size_t)e * HID + hd * HD;
    float* dst = nf + (size_t)n * HID + hd * HD;
    #pragma unroll
    for (int j = 0; j < HD; ++j) atomicAdd(&dst[j], attn * src[j]);
}

// ---------------------------------------- residual + LayerNorm, wave/node
__global__ void __launch_bounds__(256)
hg_ln_kernel(const float* __restrict__ nf,
             const float* __restrict__ hprev, int hstride,
             const float* __restrict__ g, const float* __restrict__ b,
             float* __restrict__ out, int ostride, int N)
{
    const int lane = threadIdx.x & 31;
    const int wave = threadIdx.x >> 5;
    const int node = blockIdx.x * 8 + wave;
    if (node >= N) return;

    const int c0 = lane * 4;
    const float* nr = nf + (size_t)node * HID;
    const float* hr = hprev + (size_t)node * hstride;
    float v[4];
    #pragma unroll
    for (int j = 0; j < 4; ++j) v[j] = nr[c0 + j] + hr[c0 + j];  // residual

    float sum = v[0] + v[1] + v[2] + v[3];
    #pragma unroll
    for (int o = 16; o; o >>= 1) sum += __shfl_xor(sum, o, 32);
    const float mu = sum * (1.0f / HID);

    float var = 0.0f;
    #pragma unroll
    for (int j = 0; j < 4; ++j) { const float d = v[j] - mu; var += d * d; }
    #pragma unroll
    for (int o = 16; o; o >>= 1) var += __shfl_xor(var, o, 32);
    const float inv = rsqrtf(var * (1.0f / HID) + 1e-5f);

    float* orow = out + (size_t)node * ostride;
    #pragma unroll
    for (int j = 0; j < 4; ++j)
        orow[c0 + j] = (v[j] - mu) * inv * g[c0 + j] + b[c0 + j];
}

// ---------------------------------------------------------------------------
extern "C" void kernel_launch(void* const* d_in, const int* in_sizes, int n_in,
                              void* d_out, int out_size, void* d_ws, size_t ws_size,
                              hipStream_t stream)
{
    const float* x          = (const float*)d_in[0];
    const int*   node_types = (const int*)  d_in[1];
    const int*   node_idx   = (const int*)  d_in[2];
    const int*   edge_idx   = (const int*)  d_in[3];
    const int*   edge_type  = (const int*)  d_in[4];
    const float* type_query = (const float*)d_in[5];
    const float* fc1_w      = (const float*)d_in[6];
    const float* fc1_b      = (const float*)d_in[7];
    const float* fc2_w      = (const float*)d_in[8];
    const float* fc2_b      = (const float*)d_in[9];
    const float* Wq         = (const float*)d_in[10];
    const float* Wv         = (const float*)d_in[11];
    const float* edge_ctx   = (const float*)d_in[12];
    const float* ln_g       = (const float*)d_in[13];
    const float* ln_b       = (const float*)d_in[14];

    const int  N    = in_sizes[1];
    const long NNZ  = in_sizes[2];
    const int  E    = in_sizes[4];
    const int  OST  = HID * (1 + NLAYERS);     // 384 output row stride

    // workspace layout (floats); nf/ef/s contiguous for one zero-fill
    float* ws = (float*)d_ws;
    size_t o = 0;
    float* Qb    = ws + o; o += (size_t)N * HID;
    float* Vb    = ws + o; o += (size_t)N * HID;
    float* alpha = ws + o; o += (size_t)N * HEADS;
    float* exb   = ws + o; o += (size_t)NNZ * HEADS;
    float* nf    = ws + o; o += (size_t)N * HID;     // node accumulator
    float* ef    = ws + o; o += (size_t)E * HID;     // edge accumulator
    float* sb    = ws + o; o += (size_t)E * HEADS;   // softmax denominators
    const long zcount = (size_t)N * HID + (size_t)E * HID + (size_t)E * HEADS;

    float* out = (float*)d_out;

    // stats[0] = x  -> output columns [0,128)
    {
        long tot = (long)N * HID;
        hg_copy_x_kernel<<<(int)((tot + 255) / 256), 256, 0, stream>>>(x, out, OST, N);
    }

    for (int l = 0; l < NLAYERS; ++l) {
        const float* hsrc = out + (size_t)l * HID;        // cols [128l, 128l+128)
        float*       hdst = out + (size_t)(l + 1) * HID;

        hg_zero_kernel<<<2048, 256, 0, stream>>>(nf, zcount);

        hg_qv_wmma_kernel<<<dim3((N + 15) / 16, 4), 128, 0, stream>>>(
            hsrc, OST,
            Wq + (size_t)l * HEADS * INDIM * HD,
            Wv + (size_t)l * HEADS * INDIM * HD,
            Qb, Vb, N);

        hg_mlp_alpha_wmma_kernel<<<(N + 15) / 16, 128, 0, stream>>>(
            hsrc, OST, node_types,
            type_query + (size_t)l * HEADS * NT * QD,
            fc1_w + (size_t)l * HEADS * KMLP * 32,
            fc1_b + (size_t)l * HEADS * 32,
            fc2_w + (size_t)l * HEADS * 32,
            fc2_b + (size_t)l * HEADS,
            alpha, N);

        hg_attn_ex_kernel<<<(int)((NNZ + 255) / 256), 256, 0, stream>>>(
            Qb, alpha, node_idx, edge_idx, edge_type,
            edge_ctx + (size_t)l * HEADS * ET * HD,
            exb, sb, NNZ);

        hg_edge_accum_kernel<<<(int)((NNZ * HEADS + 255) / 256), 256, 0, stream>>>(
            exb, Vb, node_idx, edge_idx, ef, NNZ);

        hg_edge_scale_kernel<<<(E * HID + 255) / 256, 256, 0, stream>>>(ef, sb, E);

        hg_node_accum_kernel<<<(int)((NNZ * HEADS + 255) / 256), 256, 0, stream>>>(
            exb, sb, ef, node_idx, edge_idx, nf, NNZ);

        hg_ln_kernel<<<(N + 7) / 8, 256, 0, stream>>>(
            nf, hsrc, OST, ln_g + (size_t)l * HID, ln_b + (size_t)l * HID,
            hdst, OST, N);
    }
}
